// GCN_3195455668886
// MI455X (gfx1250) — compile-verified, matrix operand order checked by reading, not run on hardware
//
#include <hip/hip_runtime.h>

#define N_NODES 100000
#define D_IN 256
#define D_H 128
#define D_OUT 32

typedef __attribute__((ext_vector_type(16))) __bf16 v16bf;
typedef __attribute__((ext_vector_type(8)))  float  v8f;

// Build a 16-element bf16 WMMA fragment from an LDS tile stored with 32
// bf16 per row. Per ISA 7.12.2 (16-bit A 16x32): lane<16 -> K {0..7,16..23},
// lane>=16 -> K {8..15,24..31}. Two aligned 16B LDS reads.
__device__ __forceinline__ v16bf frag_from_lds(const __bf16* base, int rowIdx, int khalf) {
  union { uint4 u[2]; v16bf v; } c;
  const uint4* p = (const uint4*)(base + rowIdx * 32 + khalf);
  c.u[0] = p[0];   // K khalf .. khalf+7
  c.u[1] = p[2];   // K khalf+16 .. khalf+23
  return c.v;
}

// C[M,N] = A[M,K] @ W[K,N], f32 in/out, bf16 WMMA math, f32 accumulate.
// 128 rows per block, 8 waves, wave w owns rows [w*16, w*16+16) x all N.
// Fused epilogue: out2[i,f] = C[i,f] * dinv[i]^2 + bias[f]  (GCN self-loop
// message + bias = init value of the scatter-add destination).
template<int K, int N>
__global__ __launch_bounds__(256) void gemm_bf16_wmma(
    const float* __restrict__ A, const float* __restrict__ W,
    float* __restrict__ C,
    const float* __restrict__ dinv, const float* __restrict__ bias,
    float* __restrict__ out2, int M) {
  constexpr int NT = N / 16;
  __shared__ __bf16 ldsA[128 * 32];   // [row][k]
  __shared__ __bf16 ldsB[N * 32];     // [n][k]  (column-major W tile)

  const int tid   = threadIdx.x;
  const int wave  = tid >> 5;
  const int lane  = tid & 31;
  const int laneM = lane & 15;
  const int khalf = (lane >> 4) << 3;
  const int rowBlk = blockIdx.x * 128;

  // A-tile fill assignment: thread owns 16 contiguous elements of one row
  const int ldRow = tid >> 1;            // 0..127
  const int ldC0  = (tid & 1) << 4;      // 0 or 16
  const int ldGr  = rowBlk + ldRow;

  v8f acc[NT];
#pragma unroll
  for (int t = 0; t < NT; ++t) {
    v8f z = {0.f, 0.f, 0.f, 0.f, 0.f, 0.f, 0.f, 0.f};
    acc[t] = z;
  }

  for (int kk = 0; kk < K; kk += 32) {
    __syncthreads();

    // ---- A tile 128x32: 4x global_load_b128 + 2x ds_store_b128 per thread
    {
      union { __bf16 h[16]; uint4 q[2]; } buf;
      if (ldGr < M) {
        const float4* src = (const float4*)(A + (size_t)ldGr * K + kk + ldC0);
#pragma unroll
        for (int j = 0; j < 4; ++j) {
          float4 v = src[j];
          buf.h[j * 4 + 0] = (__bf16)v.x;
          buf.h[j * 4 + 1] = (__bf16)v.y;
          buf.h[j * 4 + 2] = (__bf16)v.z;
          buf.h[j * 4 + 3] = (__bf16)v.w;
        }
      } else {
#pragma unroll
        for (int j = 0; j < 16; ++j) buf.h[j] = (__bf16)0.0f;
      }
      uint4* dst = (uint4*)(ldsA + ldRow * 32 + ldC0);
      dst[0] = buf.q[0];
      dst[1] = buf.q[1];
    }

    // ---- W tile 32xN -> ldsB[n][k]: thread converts one k-quad of one
    // column. Global reads coalesced over n; one ds_store_b64 per quad.
#pragma unroll
    for (int i = tid; i < N * 8; i += 256) {
      const int n  = i % N;
      const int kq = i / N;              // k = kq*4 .. kq*4+3
      union { __bf16 h[4]; uint2 q; } wb;
#pragma unroll
      for (int j = 0; j < 4; ++j)
        wb.h[j] = (__bf16)W[(size_t)(kk + kq * 4 + j) * N + n];
      *(uint2*)(ldsB + n * 32 + kq * 4) = wb.q;
    }
    __syncthreads();

    // ---- math: one A frag, NT WMMAs, B frags pipelined one tile ahead
    v16bf a = frag_from_lds(ldsA, (wave << 4) + laneM, khalf);
    v16bf bc = frag_from_lds(ldsB, laneM, khalf);
#pragma unroll
    for (int t = 0; t < NT; ++t) {
      v16bf bn = (t + 1 < NT) ? frag_from_lds(ldsB, ((t + 1) << 4) + laneM, khalf) : bc;
      acc[t] = __builtin_amdgcn_wmma_f32_16x16x32_bf16(
          false, a, false, bc, (short)0, acc[t], false, false);
      bc = bn;
    }
  }

  // Preload the bias fragment this lane needs (col = t*16 + laneM).
  float bfrag[NT];
#pragma unroll
  for (int t = 0; t < NT; ++t) bfrag[t] = bias[(t << 4) + laneM];

  // C/D layout: VGPR v, lanes 0-15 -> M=v, lanes 16-31 -> M=v+8, col=lane&15
  const int mBase = rowBlk + (wave << 4) + ((lane >> 4) << 3);
#pragma unroll
  for (int v = 0; v < 8; ++v) {
    const int gr = mBase + v;
    if (gr < M) {
      const float d  = dinv[gr];
      const float dd = d * d;
#pragma unroll
      for (int t = 0; t < NT; ++t) {
        const size_t idx = (size_t)gr * N + (t << 4) + laneM;
        const float val = acc[t][v];
        C[idx]    = val;
        out2[idx] = val * dd + bfrag[t];
      }
    }
  }
}

__global__ void k_fill1(float* __restrict__ p, int n) {
  int i = blockIdx.x * blockDim.x + threadIdx.x;
  if (i < n) p[i] = 1.0f;   // self-loop contributes 1 to every degree
}

__global__ void k_deg(const int* __restrict__ col, float* __restrict__ deg, int E) {
  int e = blockIdx.x * blockDim.x + threadIdx.x;
  if (e < E) atomicAdd(&deg[col[e]], 1.0f);
}

__global__ void k_rsqrt(float* __restrict__ p, int n) {
  int i = blockIdx.x * blockDim.x + threadIdx.x;
  if (i < n) p[i] = rsqrtf(p[i]);   // deg -> deg^{-1/2} in place
}

// Scatter-add edge messages: out[col] += h[row] * dinv[row]*dinv[col].
// Thread (edge-slot, feature) layout -> coalesced gather + coalesced atomics.
template<int F>
__global__ __launch_bounds__(256) void k_edges(
    const float* __restrict__ h, const float* __restrict__ dinv,
    const int* __restrict__ row, const int* __restrict__ col,
    float* __restrict__ out, int E) {
  constexpr int EPB = 256 / F;
  const int f  = threadIdx.x & (F - 1);
  const int es = threadIdx.x / F;
  for (int e = blockIdx.x * EPB + es; e < E; e += gridDim.x * EPB) {
    int r = row[e], c = col[e];
    float nrm = dinv[r] * dinv[c];
    atomicAdd(&out[(size_t)c * F + f], h[(size_t)r * F + f] * nrm);
  }
}

__global__ void k_relu4(float4* __restrict__ p, int n4) {
  int i = blockIdx.x * blockDim.x + threadIdx.x;
  if (i < n4) {
    float4 v = p[i];
    v.x = fmaxf(v.x, 0.0f);
    v.y = fmaxf(v.y, 0.0f);
    v.z = fmaxf(v.z, 0.0f);
    v.w = fmaxf(v.w, 0.0f);
    p[i] = v;
  }
}

// 32 classes == wave32: one wave per node, shuffle reductions.
__global__ __launch_bounds__(256) void k_log_softmax32(float* __restrict__ out, int n) {
  int node = blockIdx.x * 8 + (threadIdx.x >> 5);
  int lane = threadIdx.x & 31;
  if (node >= n) return;
  float v = out[node * 32 + lane];
  float m = v;
#pragma unroll
  for (int o = 16; o > 0; o >>= 1) m = fmaxf(m, __shfl_xor(m, o, 32));
  float ex = __expf(v - m);
  float s = ex;
#pragma unroll
  for (int o = 16; o > 0; o >>= 1) s += __shfl_xor(s, o, 32);
  out[node * 32 + lane] = v - m - __logf(s);
}

extern "C" void kernel_launch(void* const* d_in, const int* in_sizes, int n_in,
                              void* d_out, int out_size, void* d_ws, size_t ws_size,
                              hipStream_t stream) {
  const float* x  = (const float*)d_in[0];
  const float* W1 = (const float*)d_in[1];
  const float* b1 = (const float*)d_in[2];
  const float* W2 = (const float*)d_in[3];
  const float* b2 = (const float*)d_in[4];
  const int*  eix = (const int*)d_in[5];

  const int n = N_NODES;
  const int E = in_sizes[5] / 2;       // edge_index is [2, E] flat
  const int* row = eix;                // sources
  const int* col = eix + E;            // targets

  float* ws   = (float*)d_ws;
  float* dinv = ws;                                  // n floats (padded region)
  float* h    = ws + 100352;                         // n * D_H
  float* h1   = h  + (size_t)n * D_H;                // n * D_H
  float* g2   = h1 + (size_t)n * D_H;                // n * D_OUT
  float* outf = (float*)d_out;                       // n * D_OUT

  // degree / normalization
  k_fill1 <<<(n + 255) / 256, 256, 0, stream>>>(dinv, n);
  k_deg   <<<(E + 255) / 256, 256, 0, stream>>>(col, dinv, E);
  k_rsqrt <<<(n + 255) / 256, 256, 0, stream>>>(dinv, n);

  // layer 1: h = x @ W1 (epilogue: h1 = h*dinv^2 + b1) ; edge scatter ; relu
  gemm_bf16_wmma<D_IN, D_H><<<(n + 127) / 128, 256, 0, stream>>>(
      x, W1, h, dinv, b1, h1, n);
  k_edges<D_H><<<4096, 256, 0, stream>>>(h, dinv, row, col, h1, E);
  k_relu4<<<((n * (D_H / 4)) + 255) / 256, 256, 0, stream>>>((float4*)h1, n * (D_H / 4));

  // layer 2: g2 = h1 @ W2 (epilogue: out = g2*dinv^2 + b2) ; edge scatter ;
  // log_softmax in place on d_out
  gemm_bf16_wmma<D_H, D_OUT><<<(n + 127) / 128, 256, 0, stream>>>(
      h1, W2, g2, dinv, b2, outf, n);
  k_edges<D_OUT><<<4096, 256, 0, stream>>>(g2, dinv, row, col, outf, E);
  k_log_softmax32<<<(n + 7) / 8, 256, 0, stream>>>(outf, n);
}
